// QueryTopKPropagation_5875515261422
// MI455X (gfx1250) — matrix operand order repaired and checked by reading, not compiled, not run on hardware
//
#include <hip/hip_runtime.h>
#include <hip/hip_bf16.h>

// ---------------------------------------------------------------------------
// QueryTopKPropagation for MI455X (gfx1250, wave32, WMMA)
//   B=1, Q=4096, S=32768, D=256, TOPK=32
// Pass 0: one-shot fp32 -> bf16 conversion of source_val into workspace
//         (removes 8 v_cvt_pk per WMMA from the hot loop, halves B bytes)
// Pass 1: bf16 WMMA score GEMM streamed over S with per-query running top-32
//         (4 S-partitions for parallelism; partial top-k into workspace)
// Pass 2: merge 4x32 candidates -> global top-32, softsign edges,
//         weighted gather-reduce for delta_state / delta_val.
// ---------------------------------------------------------------------------

#define QX     4096
#define SX     32768
#define DK     256
#define KTOP   32
#define PARTS  4
#define SPART  (SX / PARTS)     // 8192
#define NB     32               // source columns per streaming chunk
#define WAVES  8
#define QPW    16               // queries per wave
#define QPB    (WAVES * QPW)    // 128 queries per block

typedef __attribute__((ext_vector_type(16))) __bf16 v16bf;
typedef __attribute__((ext_vector_type(8)))  __bf16 v8bf;
typedef __attribute__((ext_vector_type(8)))  float  v8f;

#define NEG_INF (-3.402823466e+38f)

// ---------------- Pass 0: fp32 -> bf16 bulk convert (8 elems/thread) -------
__global__ __launch_bounds__(256) void cvt_src_bf16(
    const float* __restrict__ in,   // [S*D]
    __bf16* __restrict__ out)       // [S*D]
{
    const size_t i = ((size_t)blockIdx.x * blockDim.x + threadIdx.x) * 8;
    const float4 f0 = *(const float4*)(in + i);
    const float4 f1 = *(const float4*)(in + i + 4);
    v8bf o;
    o[0] = (__bf16)f0.x; o[1] = (__bf16)f0.y;
    o[2] = (__bf16)f0.z; o[3] = (__bf16)f0.w;
    o[4] = (__bf16)f1.x; o[5] = (__bf16)f1.y;
    o[6] = (__bf16)f1.z; o[7] = (__bf16)f1.w;
    *(v8bf*)(out + i) = o;          // 16-byte store
}

// ---------------- Pass 1: WMMA score GEMM + streaming top-32 ---------------
// PRE=true : B fragments loaded directly as bf16 from pre-converted workspace
// PRE=false: B fragments converted fp32->bf16 on the fly (fallback)
template <bool PRE>
__global__ __launch_bounds__(256) void topk_pass1(
    const float*  __restrict__ qv,    // [Q, D] fp32
    const float*  __restrict__ sv,    // [S, D] fp32
    const __bf16* __restrict__ svb,   // [S, D] bf16 (PRE only)
    float* __restrict__ wsS,          // [Q, PARTS, KTOP]
    int*   __restrict__ wsI)          // [Q, PARTS, KTOP]
{
    // bank-conflict padding (+1) on 32-float rows
    __shared__ float tile[WAVES][QPW][NB + 1];
    __shared__ float topS[WAVES][QPW][KTOP + 1];
    __shared__ int   topI[WAVES][QPW][KTOP + 1];

    const int tid  = threadIdx.x;
    const int w    = tid >> 5;
    const int lane = tid & 31;
    const int m    = lane & 15;        // row (A) / col (B,C) index
    const int hi   = lane >> 4;        // lane half selects K range
    const int part = blockIdx.y;
    const int qbase = blockIdx.x * QPB + w * QPW;

    // ---- preload A fragments: 16 queries x 256 dims as 8 bf16 K-tiles ----
    // A 16x32 bf16, this code's k order (as B): lane m = M; VGPR v holds K pair
    //   kk = hi*16 + v*2
    v16bf aF[8];
#pragma unroll
    for (int kt = 0; kt < 8; ++kt) {
#pragma unroll
        for (int v = 0; v < 8; ++v) {
            const int kk = hi * 16 + v * 2;
            const float2 f = *(const float2*)(qv + (size_t)(qbase + m) * DK + kt * 32 + kk);
            aF[kt][2 * v]     = (__bf16)f.x;
            aF[kt][2 * v + 1] = (__bf16)f.y;
        }
    }

    // ---- init per-query top-k state ----
    for (int i = lane; i < QPW * (KTOP + 1); i += 32) {
        ((float*)&topS[w][0][0])[i] = NEG_INF;
        ((int*)&topI[w][0][0])[i]   = 0;
    }
    float curMin = NEG_INF;
    int   minPos = 0;

    // ---- stream this wave's S-partition ----
    const int sstart = part * SPART;
    for (int it = 0; it < SPART / NB; ++it) {
        const int sbase = sstart + it * NB;

        // prefetch next chunk's source rows into cache hierarchy
        if (it + 1 < SPART / NB) {
            if (PRE) {
                __builtin_prefetch(svb + (size_t)(sbase + NB + lane) * DK, 0, 1);
            } else {
                __builtin_prefetch(sv + (size_t)(sbase + NB + lane) * DK, 0, 1);
            }
        }

#pragma unroll
        for (int ct = 0; ct < NB / 16; ++ct) {
            const int scol = sbase + ct * 16;
            v8f acc = {};
#pragma unroll
            for (int kt = 0; kt < 8; ++kt) {
                // B 32x16 bf16, this code's k order (as A): lane m = N (source), lane half = K half,
                // 16 contiguous dims [kt*32 + hi*16 .. +15] -> one 32B bf16 load
                v16bf bF;
                if (PRE) {
                    bF = *(const v16bf*)(svb + (size_t)(scol + m) * DK + kt * 32 + hi * 16);
                } else {
                    const float* bp = sv + (size_t)(scol + m) * DK + kt * 32 + hi * 16;
#pragma unroll
                    for (int v = 0; v < 8; ++v) {
                        const float2 f = *(const float2*)(bp + 2 * v);
                        bF[2 * v]     = (__bf16)f.x;
                        bF[2 * v + 1] = (__bf16)f.y;
                    }
                }
                acc = __builtin_amdgcn_wmma_f32_16x16x32_bf16(
                          false, aF[kt], false, bF, (short)0, acc, false, false);
            }
            // C layout: lane m = N; VGPR r -> M = r + hi*8
#pragma unroll
            for (int r = 0; r < 8; ++r) {
                tile[w][r + hi * 8][ct * 16 + m] = acc[r];
            }
        }
        __builtin_amdgcn_wave_barrier();   // DS ops are in-order per wave

        // ---- running top-32 maintenance: lane q scans its row of the tile ----
        if (lane < QPW) {
            const int qrow = lane;
            for (int j = 0; j < NB; ++j) {
                const float s = tile[w][qrow][j];
                if (s > curMin) {
                    topS[w][qrow][minPos] = s;
                    topI[w][qrow][minPos] = sbase + j;
                    // recompute running min of the 32 kept entries
                    float mn = topS[w][qrow][0];
                    int   mp = 0;
                    for (int k = 1; k < KTOP; ++k) {
                        const float t = topS[w][qrow][k];
                        if (t < mn) { mn = t; mp = k; }
                    }
                    curMin = mn;
                    minPos = mp;
                }
            }
        }
        __builtin_amdgcn_wave_barrier();
    }

    // ---- write partition-local top-k to workspace ----
    if (lane < QPW) {
        const int qg = qbase + lane;
        for (int k = 0; k < KTOP; ++k) {
            wsS[((size_t)qg * PARTS + part) * KTOP + k] = topS[w][lane][k];
            wsI[((size_t)qg * PARTS + part) * KTOP + k] = topI[w][lane][k];
        }
    }
}

// ---------------- Pass 2: merge partitions + gather-reduce -----------------
__global__ __launch_bounds__(256) void topk_pass2(
    const float* __restrict__ wsS,
    const int*   __restrict__ wsI,
    const float* __restrict__ sv,       // [S, D]
    const float* __restrict__ state,    // [S]
    float* __restrict__ outState,       // [Q]
    float* __restrict__ outVal)         // [Q, D]
{
    __shared__ float cS[PARTS * KTOP];
    __shared__ int   cI[PARTS * KTOP];
    __shared__ float edges[KTOP];
    __shared__ int   sel[KTOP];

    const int q   = blockIdx.x;
    const int tid = threadIdx.x;

    if (tid < PARTS * KTOP) {
        cS[tid] = wsS[(size_t)q * PARTS * KTOP + tid];
        cI[tid] = wsI[(size_t)q * PARTS * KTOP + tid];
    }
    __syncthreads();

    // wave 0: 32 rounds of argmax over the 128 candidates
    if (tid < 32) {
        for (int k = 0; k < KTOP; ++k) {
            float best = cS[tid];
            int   bpos = tid;
#pragma unroll
            for (int s = 32; s < 128; s += 32) {
                const float t = cS[tid + s];
                if (t > best || (t == best && (tid + s) < bpos)) { best = t; bpos = tid + s; }
            }
#pragma unroll
            for (int off = 16; off >= 1; off >>= 1) {
                const float ov = __shfl_xor(best, off, 32);
                const int   op = __shfl_xor(bpos, off, 32);
                if (ov > best || (ov == best && op < bpos)) { best = ov; bpos = op; }
            }
            if (tid == 0) {
                edges[k] = best / (1.0f + fabsf(best));   // softsign
                sel[k]   = cI[bpos];
                cS[bpos] = NEG_INF;
            }
            __builtin_amdgcn_wave_barrier();
        }
    }
    __syncthreads();

    // delta_val: each thread owns one of the 256 D-dims
    float acc = 0.f;
    for (int k = 0; k < KTOP; ++k) {
        acc += edges[k] * sv[(size_t)sel[k] * DK + tid];
    }
    outVal[(size_t)q * DK + tid] = acc;

    if (tid == 0) {
        float a = 0.f;
        for (int k = 0; k < KTOP; ++k) a += edges[k] * state[sel[k]];
        outState[q] = a;
    }
}

extern "C" void kernel_launch(void* const* d_in, const int* in_sizes, int n_in,
                              void* d_out, int out_size, void* d_ws, size_t ws_size,
                              hipStream_t stream) {
    (void)in_sizes; (void)n_in; (void)out_size;
    const float* qv    = (const float*)d_in[0];   // [1, 4096, 256]
    const float* sv    = (const float*)d_in[1];   // [1, 32768, 256]
    const float* state = (const float*)d_in[2];   // [1, 32768]

    float* outState = (float*)d_out;              // [1, 4096]
    float* outVal   = (float*)d_out + QX;         // [1, 4096, 256]

    const size_t needB    = (size_t)SX * DK * sizeof(__bf16);       // 16 MB
    const size_t needCand = (size_t)QX * PARTS * KTOP * 8;          //  4 MB

    dim3 g1(QX / QPB, PARTS);     // (32, 4)

    if (ws_size >= needB + needCand) {
        // preferred path: pre-convert source to bf16 once
        __bf16* wsB = (__bf16*)d_ws;
        float*  wsS = (float*)((char*)d_ws + needB);
        int*    wsI = (int*)((char*)d_ws + needB + (size_t)QX * PARTS * KTOP * 4);

        const int nthreads = (SX * DK) / 8;                         // 1M threads
        cvt_src_bf16<<<nthreads / 256, 256, 0, stream>>>(sv, wsB);
        topk_pass1<true><<<g1, 256, 0, stream>>>(qv, sv, wsB, wsS, wsI);
        topk_pass2<<<QX, 256, 0, stream>>>(wsS, wsI, sv, state, outState, outVal);
    } else {
        // fallback: convert B on the fly inside the GEMM loop
        float* wsS = (float*)d_ws;
        int*   wsI = (int*)((char*)d_ws + (size_t)QX * PARTS * KTOP * 4);

        topk_pass1<false><<<g1, 256, 0, stream>>>(qv, sv, nullptr, wsS, wsI);
        topk_pass2<<<QX, 256, 0, stream>>>(wsS, wsI, sv, state, outState, outVal);
    }
}
